// GS3DRenderer_11982958756329
// MI455X (gfx1250) — matrix operand order repaired and checked by reading, not compiled, not run on hardware
//
#include <hip/hip_runtime.h>
#include <hip/hip_bf16.h>
#include <math.h>

typedef __attribute__((ext_vector_type(16))) _Float16 v16h;
typedef __attribute__((ext_vector_type(8)))  float    v8f;
typedef __attribute__((ext_vector_type(4)))  float    v4f;

#define GS_C        1024
#define GS_NCOL     64      // 59 real output columns padded to 64 (4 x 16-col WMMA tiles)
#define GS_OUTW     62
#define GS_MAXSTEP  0.2f
#define GS_CLIPSC   0.2f

// ---------------------------------------------------------------------------
// Prep: pack the five weight heads into one column-major f16 matrix
//   Wt[n][k] = W_concat[k][n]   (n = 0..63, k = 0..1023)
// column map: [0,48) shs | [48,51) scaling | [51,54) xyz | 54 opacity |
//             [55,59) rotation | [59,64) zero pad
// plus fused bias[64].
// ---------------------------------------------------------------------------
__global__ __launch_bounds__(256) void gs_prep_kernel(
    const float* __restrict__ Wshs, const float* __restrict__ bshs,
    const float* __restrict__ Wsc,  const float* __restrict__ bsc,
    const float* __restrict__ Wxyz, const float* __restrict__ bxyz,
    const float* __restrict__ Wop,  const float* __restrict__ bop,
    const float* __restrict__ Wrot, const float* __restrict__ brot,
    _Float16* __restrict__ Wt, float* __restrict__ bias) {
  int idx = blockIdx.x * 256 + threadIdx.x;   // 64 * 1024 = 65536 total
  int n = idx >> 10;
  int k = idx & (GS_C - 1);
  float w = 0.0f, b = 0.0f;
  if (n < 48)      { w = Wshs[k * 48 + n];        b = bshs[n];      }
  else if (n < 51) { w = Wsc [k * 3 + (n - 48)];  b = bsc[n - 48];  }
  else if (n < 54) { w = Wxyz[k * 3 + (n - 51)];  b = bxyz[n - 51]; }
  else if (n == 54){ w = Wop [k];                 b = bop[0];       }
  else if (n < 59) { w = Wrot[k * 4 + (n - 55)];  b = brot[n - 55]; }
  Wt[(size_t)n * GS_C + k] = (_Float16)w;
  if (k == 0) bias[n] = b;
}

__device__ __forceinline__ float gs_sigmoid(float v) {
  return 1.0f / (1.0f + __expf(-v));
}

#define GS_WMMA(A, B, CACC) \
  __builtin_amdgcn_wmma_f32_16x16x32_f16(false, (A), false, (B), (short)0, (CACC), false, false)

// ---------------------------------------------------------------------------
// Main fused kernel: 256 threads = 8 wave32, 128 rows per block.
// Wave w handles rows [blockRow + 16w, +16), all 64 columns (4 WMMA tiles).
// K-loop is software-pipelined one chunk deep so vmem latency of the next
// chunk's A/B fragments is hidden behind the current chunk's 4 WMMAs.
// ---------------------------------------------------------------------------
__global__ __launch_bounds__(256) void gs_main_kernel(
    const float* __restrict__ x, const float* __restrict__ pts,
    const _Float16* __restrict__ Wt, const float* __restrict__ bias,
    float* __restrict__ out) {
  __shared__ float tile[8][16][GS_NCOL];   // 32 KB, one 16x64 f32 tile per wave

  const int lane = threadIdx.x & 31;
  const int wave = threadIdx.x >> 5;
  const int m    = lane & 15;          // row within tile / column within B tile
  const int hlf  = lane >> 4;          // 0 or 1 (wave half)
  const int rowBlock = blockIdx.x * 128 + wave * 16;

  // Fold the lane-half K offset into the base pointers.
  const float* __restrict__ xrow =
      x + (size_t)(rowBlock + m) * GS_C + hlf * 16;
  const _Float16* __restrict__ wbase =
      Wt + (size_t)m * GS_C + hlf * 16;

  v8f c0 = {}, c1 = {}, c2 = {}, c3 = {};

  // A-fragment loader (contiguous K order, matching B): lane holds K run
  //   [kc + hlf*16, +16)   -> 4 x b128 + 8 x cvt_pk
  auto loadA = [&](int kc) -> v16h {
    v4f f0 = *(const v4f*)(xrow + kc);
    v4f f1 = *(const v4f*)(xrow + kc + 4);
    v4f f2 = *(const v4f*)(xrow + kc + 8);
    v4f f3 = *(const v4f*)(xrow + kc + 12);
    v16h a;
#pragma unroll
    for (int i = 0; i < 4; ++i) {
      a[i]      = (_Float16)f0[i];
      a[4 + i]  = (_Float16)f1[i];
      a[8 + i]  = (_Float16)f2[i];
      a[12 + i] = (_Float16)f3[i];
    }
    return a;
  };

  // ---- prologue: chunk 0
  v16h a  = loadA(0);
  v16h b0 = *(const v16h*)(wbase);
  v16h b1 = *(const v16h*)(wbase + 16 * GS_C);
  v16h b2 = *(const v16h*)(wbase + 32 * GS_C);
  v16h b3 = *(const v16h*)(wbase + 48 * GS_C);

  // ---- pipelined steady state
  for (int kc = 32; kc < GS_C; kc += 32) {
    if (kc + 64 <= GS_C) __builtin_prefetch(xrow + kc + 32, 0, 3);
    v16h an  = loadA(kc);
    const _Float16* wp = wbase + kc;
    v16h b0n = *(const v16h*)(wp);
    v16h b1n = *(const v16h*)(wp + 16 * GS_C);
    v16h b2n = *(const v16h*)(wp + 32 * GS_C);
    v16h b3n = *(const v16h*)(wp + 48 * GS_C);

    c0 = GS_WMMA(a, b0, c0);
    c1 = GS_WMMA(a, b1, c1);
    c2 = GS_WMMA(a, b2, c2);
    c3 = GS_WMMA(a, b3, c3);

    a = an; b0 = b0n; b1 = b1n; b2 = b2n; b3 = b3n;
  }

  // ---- epilogue of pipeline: last chunk
  c0 = GS_WMMA(a, b0, c0);
  c1 = GS_WMMA(a, b1, c1);
  c2 = GS_WMMA(a, b2, c2);
  c3 = GS_WMMA(a, b3, c3);

  // ---- dump accumulators to LDS (C/D layout: VGPR r, lane -> row r + 8*hlf, col n)
  {
    float* tp = &tile[wave][0][0];
    const int rbase = hlf * 8;
#pragma unroll
    for (int r = 0; r < 8; ++r) {
      float* rp = tp + (rbase + r) * GS_NCOL;
      rp[m]      = c0[r];
      rp[16 + m] = c1[r];
      rp[32 + m] = c2[r];
      rp[48 + m] = c3[r];
    }
  }
  __syncthreads();

  // ---- epilogue: 2 threads per row, activations + packed [N,62] store
  const int tid    = threadIdx.x;
  const int rlocal = tid >> 1;          // 0..127
  const int side   = tid & 1;
  const float* rowv = &tile[rlocal >> 4][rlocal & 15][0];
  const int row = blockIdx.x * 128 + rlocal;
  float* __restrict__ o = out + (size_t)row * GS_OUTW;

  if (side == 0) {
    // shs columns 0..31 (raw linear)
#pragma unroll
    for (int j = 0; j < 32; ++j) o[j] = rowv[j] + bias[j];
  } else {
    // shs columns 32..47
#pragma unroll
    for (int j = 32; j < 48; ++j) o[j] = rowv[j] + bias[j];
    // scaling: clamp(exp(v), 0, 0.2)
#pragma unroll
    for (int i = 0; i < 3; ++i) {
      float v = __expf(rowv[48 + i] + bias[48 + i]);
      o[48 + i] = fminf(fmaxf(v, 0.0f), GS_CLIPSC);
    }
    // xyz head: offset = (sigmoid(v)-0.5)*0.2 ; xyz = pts + offset
#pragma unroll
    for (int i = 0; i < 3; ++i) {
      float s   = gs_sigmoid(rowv[51 + i] + bias[51 + i]);
      float off = (s - 0.5f) * GS_MAXSTEP;
      o[51 + i] = off;
      o[54 + i] = pts[(size_t)row * 3 + i] + off;
    }
    // opacity: sigmoid
    o[57] = gs_sigmoid(rowv[54] + bias[54]);
    // rotation: quaternion L2 normalize (eps 1e-12)
    float q0 = rowv[55] + bias[55];
    float q1 = rowv[56] + bias[56];
    float q2 = rowv[57] + bias[57];
    float q3 = rowv[58] + bias[58];
    float nrm = sqrtf(q0 * q0 + q1 * q1 + q2 * q2 + q3 * q3);
    float inv = 1.0f / fmaxf(nrm, 1e-12f);
    o[58] = q0 * inv;
    o[59] = q1 * inv;
    o[60] = q2 * inv;
    o[61] = q3 * inv;
  }
}

extern "C" void kernel_launch(void* const* d_in, const int* in_sizes, int n_in,
                              void* d_out, int out_size, void* d_ws, size_t ws_size,
                              hipStream_t stream) {
  const float* x    = (const float*)d_in[0];
  const float* pts  = (const float*)d_in[1];
  const float* Wshs = (const float*)d_in[2];
  const float* bshs = (const float*)d_in[3];
  const float* Wsc  = (const float*)d_in[4];
  const float* bsc  = (const float*)d_in[5];
  const float* Wxyz = (const float*)d_in[6];
  const float* bxyz = (const float*)d_in[7];
  const float* Wop  = (const float*)d_in[8];
  const float* bop  = (const float*)d_in[9];
  const float* Wrot = (const float*)d_in[10];
  const float* brot = (const float*)d_in[11];

  _Float16* Wt   = (_Float16*)d_ws;                                        // 64*1024 halves = 128 KB
  float*    bias = (float*)((char*)d_ws + (size_t)GS_NCOL * GS_C * sizeof(_Float16));

  const int rows = in_sizes[0] / GS_C;   // 131072

  gs_prep_kernel<<<(GS_NCOL * GS_C) / 256, 256, 0, stream>>>(
      Wshs, bshs, Wsc, bsc, Wxyz, bxyz, Wop, bop, Wrot, brot, Wt, bias);

  gs_main_kernel<<<rows / 128, 256, 0, stream>>>(x, pts, Wt, bias, (float*)d_out);
}
